// Masked_Multi_Head_Attention_Layer_11630771438410
// MI455X (gfx1250) — matrix operand order repaired and checked by reading, not compiled, not run on hardware
//
#include <hip/hip_runtime.h>
#include <hip/hip_bf16.h>

// ---------------------------------------------------------------------------
// Masked MHA for MI455X (gfx1250): WMMA f16 (f32 accum), flash-attention,
// async global->LDS staging (ASYNCcnt path).
// B=4, S=2048, D=512, H=8, DH=64.
// ---------------------------------------------------------------------------

typedef __attribute__((ext_vector_type(16))) _Float16 v16h;
typedef __attribute__((ext_vector_type(8)))  _Float16 v8h;
typedef __attribute__((ext_vector_type(8)))  float    v8f;

#define B_  4
#define S_  2048
#define D_  512
#define H_  8
#define DH_ 64
#define NEG_LARGE (-1e30f)
#define LOG2E 1.44269504088896340736f

union AF { v16h v; v8h h8[2]; _Float16 h[16]; };
union CF { v8f  v; float f[8]; };
union HF { v8h  v; _Float16 h[8]; };

__device__ __forceinline__ int lane_id() { return threadIdx.x & 31; }

// Async copy of 32 contiguous bytes (per lane) global -> LDS.
// ISA 08: the instruction offset is added to both the LDS and the global address,
//         so one base pair, offset:16 for
// the second half. Tracked by ASYNCcnt.
__device__ __forceinline__ void async_copy_b256(const _Float16* gsrc,
                                                _Float16* lds_dst) {
  unsigned lds = (unsigned)(uintptr_t)lds_dst;
  unsigned long long ga = (unsigned long long)(uintptr_t)gsrc;
  asm volatile(
      "global_load_async_to_lds_b128 %0, %1, off\n\t"
      "global_load_async_to_lds_b128 %0, %1, off offset:16"
      :
      : "v"(lds), "v"(ga)
      : "memory");
}

__device__ __forceinline__ void wait_async() {
  asm volatile("s_wait_asynccnt 0x0" ::: "memory");
}

// A-frag (16x32 f16, M rows within lane). src row-major, ld in elements.
// lane l<16: row=l, k in {0..15}; lane l>=16: row=l-16, k in {16..31} (contiguous).
__device__ __forceinline__ v16h load_matA(const _Float16* base, int ld) {
  const int l = lane_id();
  const _Float16* p = base + (size_t)(l & 15) * ld + ((l < 16) ? 0 : 16);
  AF a;
  a.h8[0] = *reinterpret_cast<const v8h*>(p);
  a.h8[1] = *reinterpret_cast<const v8h*>(p + 8);
  return a.v;
}

// B-frag (32x16 f16, K x N) loaded from B-transposed storage [N][K].
// lane l: column n = l&15; k window = (l<16) ? 0..15 : 16..31 (contiguous).
__device__ __forceinline__ v16h load_matB(const _Float16* base, int ld) {
  const int l = lane_id();
  const _Float16* p = base + (size_t)(l & 15) * ld + ((l < 16) ? 0 : 16);
  AF a;
  a.h8[0] = *reinterpret_cast<const v8h*>(p);
  a.h8[1] = *reinterpret_cast<const v8h*>(p + 8);
  return a.v;
}

__device__ __forceinline__ v8f wmma_f16(v16h a, v16h b, v8f c) {
  return __builtin_amdgcn_wmma_f32_16x16x32_f16(false, a, false, b, (short)0, c,
                                                false, false);
}

// ---------------------------------------------------------------------------
// Kernel 1: QKV projection.  [8192,512]f32 @ [512,512]f32 -> f16 head-major
// [B,H,S,DH].  Q scaled by 1/8.  grid = (64, 8, 3), block = 256 (8 waves).
// (fp32->f16 conversion is needed, so staging stays through VGPRs here.)
// ---------------------------------------------------------------------------
__global__ __launch_bounds__(256)
void qkv_proj_kernel(const float* __restrict__ x,
                     const float* __restrict__ Wq, const float* __restrict__ bq,
                     const float* __restrict__ Wk, const float* __restrict__ bk,
                     const float* __restrict__ Wv, const float* __restrict__ bv,
                     _Float16* __restrict__ Qh, _Float16* __restrict__ Kh,
                     _Float16* __restrict__ Vh) {
  const int z = blockIdx.z;
  const float* W    = (z == 0) ? Wq : (z == 1) ? Wk : Wv;
  const float* bias = (z == 0) ? bq : (z == 1) ? bk : bv;
  _Float16*   out   = (z == 0) ? Qh : (z == 1) ? Kh : Vh;
  const float scale = (z == 0) ? 0.125f : 1.0f;   // 1/sqrt(DH) folded into Q

  __shared__ _Float16 As[128 * 32];   // [row][k]  8 KB
  __shared__ _Float16 Bts[64 * 32];   // [n][k]    4 KB (W transposed tile)

  const int tid = threadIdx.x;
  const int w   = tid >> 5;
  const int l   = tid & 31;
  const int m0  = blockIdx.x * 128;
  const int n0  = blockIdx.y * 64;

  v8f acc[4] = {};

  for (int kk = 0; kk < D_; kk += 32) {
    // --- stage A (x tile, fp32 -> f16), 16 elems/thread ---
    {
      const int r  = tid >> 1;
      const int ko = (tid & 1) * 16;
      const float4* src =
          reinterpret_cast<const float4*>(x + (size_t)(m0 + r) * D_ + kk + ko);
      float4 f0 = src[0], f1 = src[1], f2 = src[2], f3 = src[3];
      v8h h0 = {(_Float16)f0.x, (_Float16)f0.y, (_Float16)f0.z, (_Float16)f0.w,
                (_Float16)f1.x, (_Float16)f1.y, (_Float16)f1.z, (_Float16)f1.w};
      v8h h1 = {(_Float16)f2.x, (_Float16)f2.y, (_Float16)f2.z, (_Float16)f2.w,
                (_Float16)f3.x, (_Float16)f3.y, (_Float16)f3.z, (_Float16)f3.w};
      *reinterpret_cast<v8h*>(&As[r * 32 + ko])     = h0;
      *reinterpret_cast<v8h*>(&As[r * 32 + ko + 8]) = h1;
      if (kk + 32 < D_)  // prefetch next x tile row into caches
        __builtin_prefetch(x + (size_t)(m0 + r) * D_ + kk + 32 + ko, 0, 3);
    }
    // --- stage Bt (W tile transposed, fp32 -> f16), 8 elems/thread ---
    {
      const int kl = tid & 31;
      const int g  = tid >> 5;
      const float* src = W + (size_t)(kk + kl) * D_ + n0 + g * 8;
#pragma unroll
      for (int i = 0; i < 8; ++i)
        Bts[(g * 8 + i) * 32 + kl] = (_Float16)src[i];
    }
    __syncthreads();

    const v16h a = load_matA(&As[(w * 16) * 32], 32);
#pragma unroll
    for (int f = 0; f < 4; ++f) {
      const v16h b = load_matB(&Bts[(f * 16) * 32], 32);
      acc[f] = wmma_f16(a, b, acc[f]);
    }
    __syncthreads();
  }

  // --- epilogue: bias, scale, scatter to head-major f16 [B,H,S,DH] ---
  const int qi   = l & 15;
  const int rofs = (l < 16) ? 0 : 8;
#pragma unroll
  for (int f = 0; f < 4; ++f) {
    const int   n  = n0 + f * 16 + qi;
    const float bv_ = bias[n];
    const int   hh = n >> 6, dh = n & 63;
    CF c; c.v = acc[f];
#pragma unroll
    for (int r = 0; r < 8; ++r) {
      const int m  = m0 + w * 16 + r + rofs;
      const int bb = m >> 11, s = m & (S_ - 1);
      out[(((size_t)(bb * H_ + hh) * S_) + s) * DH_ + dh] =
          (_Float16)((c.f[r] + bv_) * scale);
    }
  }
}

// ---------------------------------------------------------------------------
// Kernel 2: causal flash attention, head-major f16 Q/K/V -> concat f16 [B,S,D].
// grid = (S/64, B*H), block = 128 (4 waves). Wave owns 16 queries.
// K tile is async-copied to LDS once per kv-block (shared by all 4 waves);
// V tile is transposed into LDS; S^T = K @ Q^T; O^T += V^T @ P^T.
// ---------------------------------------------------------------------------
__global__ __launch_bounds__(128)
void attn_kernel(const _Float16* __restrict__ Qh, const _Float16* __restrict__ Kh,
                 const _Float16* __restrict__ Vh, _Float16* __restrict__ Hh) {
  __shared__ _Float16 Ks[32 * 64];  // [key][d]  K tile as-is, 4 KB
  __shared__ _Float16 Vt[64 * 32];  // [d][key]  transposed V tile, 4 KB

  const int tid = threadIdx.x;
  const int w   = tid >> 5;
  const int l   = tid & 31;
  const int qi  = l & 15;
  const int bh  = blockIdx.y;
  const int b   = bh >> 3;
  const int h   = bh & 7;
  const int qb  = blockIdx.x * 64;
  const int qw  = qb + w * 16;
  const int myq = qw + qi;

  const _Float16* Qbase = Qh + (size_t)bh * S_ * DH_;
  const _Float16* Kbase = Kh + (size_t)bh * S_ * DH_;
  const _Float16* Vbase = Vh + (size_t)bh * S_ * DH_;

  // Q^T B-frags for the two dh-32 steps (direct contiguous global loads)
  v16h qf[2];
#pragma unroll
  for (int s = 0; s < 2; ++s) {
    const _Float16* p =
        Qbase + (size_t)(qw + qi) * DH_ + s * 32 + ((l < 16) ? 0 : 16);
    AF a;
    a.h8[0] = *reinterpret_cast<const v8h*>(p);
    a.h8[1] = *reinterpret_cast<const v8h*>(p + 8);
    qf[s] = a.v;
  }

  v8f o0 = {}, o1 = {}, o2 = {}, o3 = {};
  float m_run = NEG_LARGE, l_run = 0.f;

  const int kend = qb + 64;  // causal upper bound for this block
  for (int kb = 0; kb < kend; kb += 32) {
    // --- async-stage K tile (straight copy, ASYNCcnt-tracked) ---
    {
      const int key = tid >> 2;
      const int off = (tid & 3) * 16;  // halves
      async_copy_b256(Kbase + (size_t)(kb + key) * DH_ + off,
                      &Ks[key * 64 + off]);
    }
    // --- stage V tile transposed: Vt[d][key] (needs transpose -> manual) ---
    {
      const int key  = tid >> 2;
      const int dblk = (tid & 3) * 16;
      const _Float16* p = Vbase + (size_t)(kb + key) * DH_ + dblk;
      HF v0, v1;
      v0.v = *reinterpret_cast<const v8h*>(p);
      v1.v = *reinterpret_cast<const v8h*>(p + 8);
#pragma unroll
      for (int i = 0; i < 8; ++i) Vt[(dblk + i) * 32 + key]     = v0.h[i];
#pragma unroll
      for (int i = 0; i < 8; ++i) Vt[(dblk + 8 + i) * 32 + key] = v1.h[i];
    }
    wait_async();
    __syncthreads();

    // --- scores^T: two 16-key tiles, accumulate over dh (2 k-steps) ---
    v8f st0 = {}, st1 = {};
#pragma unroll
    for (int s = 0; s < 2; ++s) {
      const v16h ka0 = load_matA(&Ks[0 * 16 * 64 + s * 32], 64);
      const v16h ka1 = load_matA(&Ks[1 * 16 * 64 + s * 32], 64);
      st0 = wmma_f16(ka0, qf[s], st0);
      st1 = wmma_f16(ka1, qf[s], st1);
    }

    // --- causal mask + online softmax ---
    CF s0, s1; s0.v = st0; s1.v = st1;
    const int keyofs = (l < 16) ? 0 : 8;
#pragma unroll
    for (int r = 0; r < 8; ++r) {
      const int k0 = kb + r + keyofs;
      if (k0 > myq)      s0.f[r] = NEG_LARGE;
      if (k0 + 16 > myq) s1.f[r] = NEG_LARGE;
    }
    float mx = NEG_LARGE;
#pragma unroll
    for (int r = 0; r < 8; ++r) mx = fmaxf(mx, fmaxf(s0.f[r], s1.f[r]));
    mx = fmaxf(mx, __shfl_xor(mx, 16, 32));
    const float m_new = fmaxf(m_run, mx);

    float p0[8], p1[8], lsum = 0.f;
#pragma unroll
    for (int r = 0; r < 8; ++r) {
      p0[r] = __builtin_exp2f((s0.f[r] - m_new) * LOG2E);
      p1[r] = __builtin_exp2f((s1.f[r] - m_new) * LOG2E);
      lsum += p0[r] + p1[r];
    }
    lsum += __shfl_xor(lsum, 16, 32);
    const float corr = __builtin_exp2f((m_run - m_new) * LOG2E);
    m_run = m_new;
    l_run = l_run * corr + lsum;
    o0 = o0 * corr; o1 = o1 * corr; o2 = o2 * corr; o3 = o3 * corr;

    // --- rebuild P^T into B-frag layout via half-wave exchanges ---
    AF pb;
#pragma unroll
    for (int r = 0; r < 8; ++r) {
      const float x0 = __shfl_xor(p0[r], 16, 32);
      const float x1 = __shfl_xor(p1[r], 16, 32);
      pb.h[r]     = (_Float16)((l < 16) ? p0[r] : x1);  // keys {0..7}/{16..23}
      pb.h[8 + r] = (_Float16)((l < 16) ? x0 : p1[r]);  // keys {8..15}/{24..31}
    }

    // --- O^T += V^T @ P^T over 4 d-tiles ---
    o0 = wmma_f16(load_matA(&Vt[0 * 16 * 32], 32), pb.v, o0);
    o1 = wmma_f16(load_matA(&Vt[1 * 16 * 32], 32), pb.v, o1);
    o2 = wmma_f16(load_matA(&Vt[2 * 16 * 32], 32), pb.v, o2);
    o3 = wmma_f16(load_matA(&Vt[3 * 16 * 32], 32), pb.v, o3);
    __syncthreads();
  }

  // --- epilogue: normalize, pack 8 f16, b128 stores in concat layout ---
  const float inv_l = 1.0f / l_run;
  _Float16* obase = Hh + (size_t)(b * S_ + myq) * D_ + h * DH_;
  const int rofs = (l < 16) ? 0 : 8;
  v8f oo[4] = {o0, o1, o2, o3};
#pragma unroll
  for (int dt = 0; dt < 4; ++dt) {
    CF c; c.v = oo[dt];
    HF hv;
#pragma unroll
    for (int r = 0; r < 8; ++r) hv.h[r] = (_Float16)(c.f[r] * inv_l);
    *reinterpret_cast<v8h*>(obase + dt * 16 + rofs) = hv.v;
  }
}

// ---------------------------------------------------------------------------
// Kernel 3: output projection.  Hh f16 [8192,512] @ Wo f32 [512,512] + bo
// -> f32 out.  A tile staged via async global->LDS copies.
// grid = (64, 8), block = 256.
// ---------------------------------------------------------------------------
__global__ __launch_bounds__(256)
void out_proj_kernel(const _Float16* __restrict__ Hh,
                     const float* __restrict__ Wo, const float* __restrict__ bo,
                     float* __restrict__ out) {
  __shared__ _Float16 As[128 * 32];
  __shared__ _Float16 Bts[64 * 32];

  const int tid = threadIdx.x;
  const int w   = tid >> 5;
  const int l   = tid & 31;
  const int m0  = blockIdx.x * 128;
  const int n0  = blockIdx.y * 64;

  v8f acc[4] = {};

  for (int kk = 0; kk < D_; kk += 32) {
    {  // stage A (already f16): async global -> LDS, no VGPR round-trip
      const int r  = tid >> 1;
      const int ko = (tid & 1) * 16;
      async_copy_b256(Hh + (size_t)(m0 + r) * D_ + kk + ko,
                      &As[r * 32 + ko]);
    }
    {  // stage Bt (Wo transposed tile, fp32 -> f16)
      const int kl = tid & 31;
      const int g  = tid >> 5;
      const float* src = Wo + (size_t)(kk + kl) * D_ + n0 + g * 8;
#pragma unroll
      for (int i = 0; i < 8; ++i)
        Bts[(g * 8 + i) * 32 + kl] = (_Float16)src[i];
    }
    wait_async();
    __syncthreads();

    const v16h a = load_matA(&As[(w * 16) * 32], 32);
#pragma unroll
    for (int f = 0; f < 4; ++f) {
      const v16h b = load_matB(&Bts[(f * 16) * 32], 32);
      acc[f] = wmma_f16(a, b, acc[f]);
    }
    __syncthreads();
  }

  const int qi   = l & 15;
  const int rofs = (l < 16) ? 0 : 8;
#pragma unroll
  for (int f = 0; f < 4; ++f) {
    const int   n  = n0 + f * 16 + qi;
    const float bv_ = bo[n];
    CF c; c.v = acc[f];
#pragma unroll
    for (int r = 0; r < 8; ++r) {
      const int m = m0 + w * 16 + r + rofs;
      out[(size_t)m * D_ + n] = c.f[r] + bv_;
    }
  }
}

// ---------------------------------------------------------------------------
extern "C" void kernel_launch(void* const* d_in, const int* in_sizes, int n_in,
                              void* d_out, int out_size, void* d_ws,
                              size_t ws_size, hipStream_t stream) {
  const float* x  = (const float*)d_in[0];
  // d_in[1] = mask (causality recomputed in-kernel; unused)
  const float* Wq = (const float*)d_in[2];
  const float* bq = (const float*)d_in[3];
  const float* Wk = (const float*)d_in[4];
  const float* bk = (const float*)d_in[5];
  const float* Wv = (const float*)d_in[6];
  const float* bv = (const float*)d_in[7];
  const float* Wo = (const float*)d_in[8];
  const float* bo = (const float*)d_in[9];
  float* out = (float*)d_out;

  const size_t elems = (size_t)B_ * S_ * D_;  // 4,194,304
  _Float16* Qh = (_Float16*)d_ws;             // 8 MB each
  _Float16* Kh = Qh + elems;
  _Float16* Vh = Kh + elems;
  _Float16* Hh = Vh + elems;

  qkv_proj_kernel<<<dim3(B_ * S_ / 128, D_ / 64, 3), 256, 0, stream>>>(
      x, Wq, bq, Wk, bk, Wv, bv, Qh, Kh, Vh);
  attn_kernel<<<dim3(S_ / 64, B_ * H_), 128, 0, stream>>>(Qh, Kh, Vh, Hh);
  out_proj_kernel<<<dim3(B_ * S_ / 128, D_ / 64), 256, 0, stream>>>(Hh, Wo, bo,
                                                                    out);
}
